// RecurrentMoE_17411797418534
// MI455X (gfx1250) — matrix operand order repaired and checked by reading, not compiled, not run on hardware
//
#include <hip/hip_runtime.h>

typedef __bf16 bf16_t;
typedef __attribute__((ext_vector_type(16))) __bf16 v16bf;
typedef __attribute__((ext_vector_type(8)))  float  v8f;
typedef __attribute__((ext_vector_type(4)))  unsigned int v4u;

union AFrag { v16bf v; v4u q[2]; };

__device__ __forceinline__ v8f wmma_bf16(const AFrag& a, const AFrag& b, v8f c) {
    return __builtin_amdgcn_wmma_f32_16x16x32_bf16(
        /*neg_a=*/false, a.v, /*neg_b=*/false, b.v,
        /*c_mod=*/(short)0, c, /*reuse_a=*/false, /*reuse_b=*/false);
}

// ---------------------------------------------------------------------------
// Elementwise helpers
// ---------------------------------------------------------------------------
__global__ void cvt_f32_bf16(bf16_t* __restrict__ dst, const float* __restrict__ src, size_t n) {
    size_t i = (size_t)blockIdx.x * blockDim.x + threadIdx.x;
    size_t stride = (size_t)gridDim.x * blockDim.x;
    for (; i < n; i += stride) dst[i] = (bf16_t)src[i];
}

__global__ void zero_f32(float* __restrict__ dst, size_t n) {
    size_t i = (size_t)blockIdx.x * blockDim.x + threadIdx.x;
    size_t stride = (size_t)gridDim.x * blockDim.x;
    for (; i < n; i += stride) dst[i] = 0.0f;
}

// ---------------------------------------------------------------------------
// Tiled WMMA GEMM:  C[M,N] = A[M,K] (bf16) * W[N,K]^T (bf16) + bias
// MODE 0: Cf = result (f32)
// MODE 1: Cb = bf16(relu(result))
// MODE 2: Cf (+)= gate[row*8+expert] * result   (overwrite when first!=0)
// 256 threads, 8 waves; block tile 128x128x32, wave tile 32x64.
// Double-buffered LDS with register staging: one barrier per K-step; global
// loads for tile k+1 issue before the WMMAs of tile k and store into the
// buffer nobody reads this iteration.
// All of M,N,K are exact multiples of the tile dims (no bounds checks).
// ---------------------------------------------------------------------------
template<int MODE>
__global__ __launch_bounds__(256)
void gemm_wmma(const bf16_t* __restrict__ A, const bf16_t* __restrict__ W,
               const float* __restrict__ bias,
               float* __restrict__ Cf, bf16_t* __restrict__ Cb,
               const float* __restrict__ gate,
               int M, int N, int K, int expert, int first) {
    constexpr int BM = 128, BN = 128, BK = 32, BKP = 56; // padded LDS stride (112B, 16B aligned)
    __shared__ bf16_t As[2][BM][BKP];
    __shared__ bf16_t Bs[2][BN][BKP];

    const int tid  = threadIdx.x;
    const int wid  = tid >> 5;
    const int lane = tid & 31;
    const int m0 = blockIdx.y * BM;
    const int n0 = blockIdx.x * BN;
    const int wm0 = (wid >> 1) * 32;   // wave row offset (0,32,64,96)
    const int wn0 = (wid & 1) * 64;    // wave col offset (0,64)

    v8f acc[2][4];
    for (int i = 0; i < 2; ++i)
        for (int j = 0; j < 4; ++j)
            for (int r = 0; r < 8; ++r) acc[i][j][r] = 0.0f;

    const int lrow = tid >> 2;         // 0..63
    const int lseg = (tid & 3) * 8;    // element offset 0,8,16,24 (16B chunks)
    const int am = wm0 + (lane & 15);
    const int ak = (lane >> 4) * 8;    // A: K-interleave of 8 across lane halves
    const int bn = wn0 + (lane & 15);
    const int bk = (lane >> 4) * 8;    // B: K-interleave of 8 across lane halves

    const int ktiles = K / BK;

    // prologue: stage tile 0 into buffer 0
    {
        for (int r = 0; r < 2; ++r) {
            const int row = lrow + r * 64;
            *(v4u*)(&As[0][row][lseg]) = *(const v4u*)(A + (size_t)(m0 + row) * K + lseg);
            *(v4u*)(&Bs[0][row][lseg]) = *(const v4u*)(W + (size_t)(n0 + row) * K + lseg);
        }
    }
    __syncthreads();

    for (int kt = 0; kt < ktiles; ++kt) {
        const int cur = kt & 1;
        const bool more = (kt + 1) < ktiles;

        // stage next tile in registers (latency hidden under the WMMAs below)
        v4u na[2], nb[2];
        if (more) {
            const int k0 = (kt + 1) * BK;
            for (int r = 0; r < 2; ++r) {
                const int row = lrow + r * 64;
                na[r] = *(const v4u*)(A + (size_t)(m0 + row) * K + k0 + lseg);
                nb[r] = *(const v4u*)(W + (size_t)(n0 + row) * K + k0 + lseg);
            }
        }

        // compute from the current buffer
        AFrag a[2], b[4];
        for (int i = 0; i < 2; ++i) {
            a[i].q[0] = *(const v4u*)(&As[cur][am + i * 16][ak]);
            a[i].q[1] = *(const v4u*)(&As[cur][am + i * 16][ak + 16]);
        }
        for (int j = 0; j < 4; ++j) {
            b[j].q[0] = *(const v4u*)(&Bs[cur][bn + j * 16][bk]);
            b[j].q[1] = *(const v4u*)(&Bs[cur][bn + j * 16][bk + 16]);
        }
        for (int i = 0; i < 2; ++i)
            for (int j = 0; j < 4; ++j)
                acc[i][j] = wmma_bf16(a[i], b[j], acc[i][j]);

        // publish the staged tile into the other buffer
        if (more) {
            const int nxt = cur ^ 1;
            for (int r = 0; r < 2; ++r) {
                const int row = lrow + r * 64;
                *(v4u*)(&As[nxt][row][lseg]) = na[r];
                *(v4u*)(&Bs[nxt][row][lseg]) = nb[r];
            }
        }
        __syncthreads();
    }

    // epilogue: VGPR r, lanes 0-15 -> row r, lanes 16-31 -> row r+8
    const int coffs = lane & 15;
    const int roffs = (lane >> 4) * 8;
    for (int i = 0; i < 2; ++i) {
        for (int j = 0; j < 4; ++j) {
            const int col = n0 + wn0 + j * 16 + coffs;
            const float bv = bias[col];
            for (int r = 0; r < 8; ++r) {
                const int row = m0 + wm0 + i * 16 + roffs + r;
                const float v = acc[i][j][r] + bv;
                if constexpr (MODE == 0) {
                    Cf[(size_t)row * N + col] = v;
                } else if constexpr (MODE == 1) {
                    Cb[(size_t)row * N + col] = (bf16_t)fmaxf(v, 0.0f);
                } else {
                    const float o = gate[row * 8 + expert] * v;
                    const size_t idx = (size_t)row * N + col;
                    if (first) Cf[idx] = o;
                    else       Cf[idx] += o;
                }
            }
        }
    }
}

// ---------------------------------------------------------------------------
// Persistent single-workgroup GRU using WMMA.
// B=4 rows ride in a zero-padded 16-row tile; 8 waves each own 18 of the
// 144 N-tiles (N=2304) for BOTH gi = x_t*Wih^T and gh = h*Whh^T per step.
// Weights are bf16 in global (L2-resident); state kept in LDS.
// ---------------------------------------------------------------------------
__global__ __launch_bounds__(256)
void gru_wmma(const float* __restrict__ h_in, float* __restrict__ h_seq,
              const bf16_t* __restrict__ Wih, const bf16_t* __restrict__ Whh,
              const float* __restrict__ bih, const float* __restrict__ bhh,
              float* __restrict__ h_state) {
    constexpr int H = 768, G = 2304, S = 512;
    constexpr int XP = H + 24;     // padded stride: 1584B, 16B aligned, bank-spread
    __shared__ bf16_t xs[16][XP];
    __shared__ bf16_t hs[16][XP];
    __shared__ float  hf[4][H];
    __shared__ float  gi[4][G];
    __shared__ float  gh[4][G];

    const int tid  = threadIdx.x;
    const int wid  = tid >> 5;
    const int lane = tid & 31;

    // init: zero both bf16 tiles (pad rows/cols), load h0 (zeros at layer 0)
    for (int idx = tid; idx < 16 * XP; idx += 256) {
        const int r = idx / XP, c = idx % XP;
        xs[r][c] = (bf16_t)0.0f;
        hs[r][c] = (bf16_t)0.0f;
    }
    __syncthreads();
    for (int idx = tid; idx < 4 * H; idx += 256) {
        const float h0 = h_state[idx];
        hf[idx / H][idx % H] = h0;
        hs[idx / H][idx % H] = (bf16_t)h0;
    }
    __syncthreads();

    const int am = lane & 15;
    const int ak = (lane >> 4) * 8;

    for (int s = 0; s < S; ++s) {
        // stage x_t (4x768 f32 -> bf16 LDS)
        for (int idx = tid; idx < 4 * H; idx += 256) {
            const int b = idx / H, j = idx % H;
            xs[b][j] = (bf16_t)h_in[((size_t)b * S + s) * H + j];
        }
        __syncthreads();

        // both GEMMs: each wave does 18 N-tiles, K=768 in 24 WMMA steps
        for (int t = 0; t < 18; ++t) {
            const int nt   = wid * 18 + t;
            const int ncol = nt * 16 + (lane & 15);
            v8f ci, ch;
            for (int r = 0; r < 8; ++r) { ci[r] = 0.0f; ch[r] = 0.0f; }
            for (int kk = 0; kk < H; kk += 32) {
                AFrag ax, ah, bi, bh;
                ax.q[0] = *(const v4u*)(&xs[am][kk + ak]);
                ax.q[1] = *(const v4u*)(&xs[am][kk + ak + 16]);
                ah.q[0] = *(const v4u*)(&hs[am][kk + ak]);
                ah.q[1] = *(const v4u*)(&hs[am][kk + ak + 16]);
                const int bk = kk + (lane >> 4) * 8;
                bi.q[0] = *(const v4u*)(Wih + (size_t)ncol * H + bk);
                bi.q[1] = *(const v4u*)(Wih + (size_t)ncol * H + bk + 16);
                bh.q[0] = *(const v4u*)(Whh + (size_t)ncol * H + bk);
                bh.q[1] = *(const v4u*)(Whh + (size_t)ncol * H + bk + 16);
                ci = wmma_bf16(ax, bi, ci);
                ch = wmma_bf16(ah, bh, ch);
            }
            if (lane < 16) {   // rows 0..3 live in VGPRs 0..3 of lanes 0..15
                for (int r = 0; r < 4; ++r) {
                    gi[r][ncol] = ci[r] + bih[ncol];
                    gh[r][ncol] = ch[r] + bhh[ncol];
                }
            }
        }
        __syncthreads();

        // elementwise GRU update (f32), write h_seq + refresh LDS state
        for (int idx = tid; idx < 4 * H; idx += 256) {
            const int b = idx / H, j = idx % H;
            const float ir = gi[b][j], iz = gi[b][H + j], in_ = gi[b][2 * H + j];
            const float hr = gh[b][j], hz = gh[b][H + j], hn  = gh[b][2 * H + j];
            const float r = 1.0f / (1.0f + __expf(-(ir + hr)));
            const float z = 1.0f / (1.0f + __expf(-(iz + hz)));
            const float n = tanhf(in_ + r * hn);
            const float hnew = (1.0f - z) * n + z * hf[b][j];
            hf[b][j] = hnew;
            hs[b][j] = (bf16_t)hnew;
            h_seq[((size_t)b * S + s) * H + j] = hnew;
        }
        __syncthreads();
    }

    for (int idx = tid; idx < 4 * H; idx += 256)
        h_state[idx] = hf[idx / H][idx % H];
}

// ---------------------------------------------------------------------------
// Router: logits = h_seq @ Wr^T + br; strict top-2 (jax first-index
// tie-break); softmax over the two; scatter into dense gate[M,8].
// ---------------------------------------------------------------------------
__global__ __launch_bounds__(256)
void router_top2(const float* __restrict__ h_seq, const float* __restrict__ Wr,
                 const float* __restrict__ br, float* __restrict__ gate) {
    constexpr int H = 768, E = 8;
    const int t = blockIdx.x * 256 + threadIdx.x;   // token 0..2047
    float acc[E];
    for (int e = 0; e < E; ++e) acc[e] = br[e];
    const float* hrow = h_seq + (size_t)t * H;
    for (int k = 0; k < H; ++k) {
        const float hv = hrow[k];
        for (int e = 0; e < E; ++e) acc[e] += hv * Wr[e * H + k];
    }
    int i0 = 0; float v0 = acc[0];
    for (int e = 1; e < E; ++e) if (acc[e] > v0) { v0 = acc[e]; i0 = e; }
    int i1 = -1; float v1 = -3.402823466e38f;
    for (int e = 0; e < E; ++e) if (e != i0 && acc[e] > v1) { v1 = acc[e]; i1 = e; }
    const float ex = __expf(v1 - v0);
    const float g0 = 1.0f / (1.0f + ex);
    const float g1 = ex / (1.0f + ex);
    float* grow = gate + (size_t)t * E;
    for (int e = 0; e < E; ++e) grow[e] = 0.0f;
    grow[i0] = g0;
    grow[i1] = g1;
}

// ---------------------------------------------------------------------------
// Host orchestration
// ---------------------------------------------------------------------------
extern "C" void kernel_launch(void* const* d_in, const int* in_sizes, int n_in,
                              void* d_out, int out_size, void* d_ws, size_t ws_size,
                              hipStream_t stream) {
    constexpr int L = 2, E = 8, B = 4, S = 512, I = 768, H = 768;
    constexpr int FF = 4 * I;      // 3072
    constexpr int G3 = 3 * H;      // 2304
    constexpr int M  = B * S;      // 2048

    const float* x_in   = (const float*)d_in[0];
    const float* proj_W = (const float*)d_in[1];
    const float* proj_b = (const float*)d_in[2];
    const float* gWih   = (const float*)d_in[3];
    const float* gWhh   = (const float*)d_in[4];
    const float* gbih   = (const float*)d_in[5];
    const float* gbhh   = (const float*)d_in[6];
    const float* rW     = (const float*)d_in[7];
    const float* rb     = (const float*)d_in[8];
    const float* eW1    = (const float*)d_in[9];
    const float* eb1    = (const float*)d_in[10];
    const float* eW2    = (const float*)d_in[11];
    const float* eb2    = (const float*)d_in[12];
    float* out = (float*)d_out;

    char* wp = (char*)d_ws;
    auto alloc = [&](size_t bytes) -> char* {
        char* p = wp;
        wp += (bytes + 255) & ~(size_t)255;
        return p;
    };
    bf16_t* projW_bf = (bf16_t*)alloc((size_t)L * H * I * 2);
    bf16_t* Wih_bf   = (bf16_t*)alloc((size_t)G3 * H * 2);
    bf16_t* Whh_bf   = (bf16_t*)alloc((size_t)G3 * H * 2);
    bf16_t* eW1_bf   = (bf16_t*)alloc((size_t)L * E * FF * I * 2);
    bf16_t* eW2_bf   = (bf16_t*)alloc((size_t)L * E * I * FF * 2);
    bf16_t* x_bf     = (bf16_t*)alloc((size_t)M * I * 2);
    bf16_t* h1_bf    = (bf16_t*)alloc((size_t)M * FF * 2);
    float*  h_in     = (float*)alloc((size_t)M * H * 4);
    float*  h_seq    = (float*)alloc((size_t)M * H * 4);
    float*  gate     = (float*)alloc((size_t)M * E * 4);
    float*  out0     = (float*)alloc((size_t)M * I * 4);
    float*  h_state  = (float*)alloc((size_t)B * H * 4);

    // weight + input conversions (f32 -> bf16), GRU state zero
    cvt_f32_bf16<<<256,  256, 0, stream>>>(projW_bf, proj_W, (size_t)L * H * I);
    cvt_f32_bf16<<<256,  256, 0, stream>>>(Wih_bf,   gWih,   (size_t)G3 * H);
    cvt_f32_bf16<<<256,  256, 0, stream>>>(Whh_bf,   gWhh,   (size_t)G3 * H);
    cvt_f32_bf16<<<2048, 256, 0, stream>>>(eW1_bf,   eW1,    (size_t)L * E * FF * I);
    cvt_f32_bf16<<<2048, 256, 0, stream>>>(eW2_bf,   eW2,    (size_t)L * E * I * FF);
    cvt_f32_bf16<<<256,  256, 0, stream>>>(x_bf,     x_in,   (size_t)M * I);
    zero_f32<<<12, 256, 0, stream>>>(h_state, (size_t)B * H);

    for (int l = 0; l < L; ++l) {
        // h_in = x @ proj_W[l]^T + proj_b[l]
        gemm_wmma<0><<<dim3(H / 128, M / 128), 256, 0, stream>>>(
            x_bf, projW_bf + (size_t)l * H * I, proj_b + (size_t)l * H,
            h_in, nullptr, nullptr, M, H, I, 0, 0);

        // GRU over the sequence (persistent, single WGP)
        gru_wmma<<<1, 256, 0, stream>>>(h_in, h_seq, Wih_bf, Whh_bf,
                                        gbih, gbhh, h_state);

        // gates
        router_top2<<<M / 256, 256, 0, stream>>>(h_seq, rW + (size_t)l * E * H,
                                                 rb + (size_t)l * E, gate);

        float* outp = (l == 0) ? out0 : out;
        for (int e = 0; e < E; ++e) {
            const size_t we = (size_t)(l * E + e);
            gemm_wmma<1><<<dim3(FF / 128, M / 128), 256, 0, stream>>>(
                x_bf, eW1_bf + we * FF * I, eb1 + we * FF,
                nullptr, h1_bf, nullptr, M, FF, I, 0, 0);
            gemm_wmma<2><<<dim3(I / 128, M / 128), 256, 0, stream>>>(
                h1_bf, eW2_bf + we * I * FF, eb2 + we * I,
                outp, nullptr, gate, M, I, FF, e, (e == 0) ? 1 : 0);
        }
        if (l == 0)
            cvt_f32_bf16<<<256, 256, 0, stream>>>(x_bf, out0, (size_t)M * I);
    }
}
